// MPNNCNN_35983236006069
// MI455X (gfx1250) — matrix operand order repaired and checked by reading, not compiled, not run on hardware
//
#include <hip/hip_runtime.h>
#include <math.h>

// ---------------- problem constants (match reference) ----------------
#define HH 384
#define WW 384
#define CC 200
#define NN (HH * WW)        // 147456, divisible by 128
#define HIDE 128
#define COUT 64
#define NCLS 16
#define SLOPE 0.01f
#define EPS 1e-5f

typedef float v8f   __attribute__((ext_vector_type(8)));
typedef __bf16 v16bf __attribute__((ext_vector_type(16)));

union Frag { v16bf v; uint4 q[2]; };

__device__ __forceinline__ __bf16 f2bf(float f) {
  union { float f; unsigned u; } in; in.f = f;
  unsigned u = in.u;
  u += 0x7fffu + ((u >> 16) & 1u);           // round-to-nearest-even
  union { unsigned short s; __bf16 b; } out; out.s = (unsigned short)(u >> 16);
  return out.b;
}

__device__ __forceinline__ float leaky(float v) { return v > 0.f ? v : SLOPE * v; }

// ---------------- small utility kernels ----------------
__global__ void zero_kernel(float* p, int n) {
  int i = blockIdx.x * blockDim.x + threadIdx.x;
  if (i < n) p[i] = 0.f;
}
__global__ void fill_kernel(float* p, float v, int n) {
  int i = blockIdx.x * blockDim.x + threadIdx.x;
  if (i < n) p[i] = v;
}
__global__ void edge_deg_kernel(float* deg, const int* __restrict__ dst, int E) {
  int e = blockIdx.x * blockDim.x + threadIdx.x;
  if (e < E) atomicAdd(&deg[dst[e]], 1.0f);
}
__global__ void rsqrt_inplace_kernel(float* p, int n) {
  int i = blockIdx.x * blockDim.x + threadIdx.x;
  if (i < n) p[i] = rsqrtf(p[i]);
}

// per-channel sum / sum-of-squares over N rows (channel-last layout)
__global__ __launch_bounds__(256)
void colstats_kernel(const float* __restrict__ in, float* __restrict__ sum,
                     float* __restrict__ sumsq, int Cn, int n) {
  int t = threadIdx.x;
  if (t >= Cn) return;
  float ls = 0.f, lq = 0.f;
  for (int r = blockIdx.x; r < n; r += gridDim.x) {
    float v = in[(size_t)r * Cn + t];
    ls += v; lq += v * v;
  }
  atomicAdd(&sum[t], ls);
  atomicAdd(&sumsq[t], lq);
}

// scale = g*rsqrt(var+eps), shift = b - mean*scale  (training-mode BN)
__global__ void finalize_affine_kernel(const float* __restrict__ sum, const float* __restrict__ sumsq,
                                       const float* __restrict__ g, const float* __restrict__ b,
                                       float* __restrict__ scale, float* __restrict__ shift,
                                       int Cn, float invN) {
  int t = threadIdx.x;
  if (t >= Cn) return;
  float m = sum[t] * invN;
  float var = sumsq[t] * invN - m * m;
  float sc = g[t] * rsqrtf(var + EPS);
  scale[t] = sc;
  shift[t] = b[t] - m * sc;
}

// row L2-normalize (width 128) + accumulate column stats of normalized output
__global__ __launch_bounds__(128)
void rownorm_stats_kernel(const float* __restrict__ in, float* __restrict__ out,
                          float* __restrict__ sum, float* __restrict__ sumsq, int n) {
  __shared__ float red[128];
  int t = threadIdx.x;
  float ls = 0.f, lq = 0.f;
  for (int r = blockIdx.x; r < n; r += gridDim.x) {
    float v = in[(size_t)r * HIDE + t];
    red[t] = v * v;
    __syncthreads();
    for (int off = 64; off > 0; off >>= 1) {
      if (t < off) red[t] += red[t + off];
      __syncthreads();
    }
    float nrm = fmaxf(sqrtf(red[0]), 1e-12f);
    __syncthreads();
    float o = v / nrm;
    out[(size_t)r * HIDE + t] = o;
    ls += o; lq += o * o;
  }
  atomicAdd(&sum[t], ls);
  atomicAdd(&sumsq[t], lq);
}

// ---------------- WMMA GEMM: Y = act( affine(X) @ W + bias ) ----------------
// - whole W preloaded to LDS as bf16 (transposed, K-contiguous per column)
// - A tile (128 x 32) double-buffered in LDS, staged with float4 loads
// - B fragments software-pipelined in registers across the NT WMMAs
template <int K_TOT, int NOUT, bool LEAKY, bool HAS_BIAS>
__global__ __launch_bounds__(256)
void gemm_bn_wmma(const float* __restrict__ X, const float* __restrict__ scale,
                  const float* __restrict__ shift, const float* __restrict__ W,
                  const float* __restrict__ bias, float* __restrict__ Y) {
  static_assert(K_TOT % 4 == 0, "K_TOT must be multiple of 4");
  constexpr int KSTEPS = (K_TOT + 31) / 32;
  constexpr int KPAD = KSTEPS * 32;
  constexpr int BSTRIDE = KPAD + 8;   // multiple of 8 bf16 -> 16B-aligned rows
  constexpr int NT = NOUT / 16;
  __shared__ __align__(16) __bf16 As[2][128][40];
  __shared__ __align__(16) __bf16 Bs[NOUT][BSTRIDE];

  const int t = threadIdx.x;
  const int wave = t >> 5;
  const int lane = t & 31;
  const int row0 = blockIdx.x * 128;
  const int mrow = (wave << 4) + (lane & 15);
  const int khalf = lane >> 4;
  const int coln = lane & 15;

  // ---- preload entire B (transposed, bf16); coalesced float4 along n ----
  for (int idx = t * 4; idx < K_TOT * NOUT; idx += 256 * 4) {
    int n = idx % NOUT;       // multiple of 4
    int kg = idx / NOUT;
    const float4 w4 = *reinterpret_cast<const float4*>(&W[(size_t)kg * NOUT + n]);
    Bs[n + 0][kg] = f2bf(w4.x);
    Bs[n + 1][kg] = f2bf(w4.y);
    Bs[n + 2][kg] = f2bf(w4.z);
    Bs[n + 3][kg] = f2bf(w4.w);
  }
  if constexpr (KPAD > K_TOT) {
    for (int idx = t; idx < NOUT * (KPAD - K_TOT); idx += 256) {
      int n = idx % NOUT;
      int kg = K_TOT + idx / NOUT;
      Bs[n][kg] = f2bf(0.f);
    }
  }

  // ---- A stager: 128x32 tile, float4 chunks (chunk-uniform K bound) ----
  auto stageA = [&](int ks, int buf) {
    const int k0 = ks * 32;
    for (int idx = t * 4; idx < 128 * 32; idx += 256 * 4) {
      int r = idx >> 5;
      int k = idx & 31;          // multiple of 4
      int kg = k0 + k;
      if (kg < K_TOT) {
        const float4 x4 = *reinterpret_cast<const float4*>(&X[(size_t)(row0 + r) * K_TOT + kg]);
        const float4 s4 = *reinterpret_cast<const float4*>(&scale[kg]);
        const float4 h4 = *reinterpret_cast<const float4*>(&shift[kg]);
        As[buf][r][k + 0] = f2bf(x4.x * s4.x + h4.x);
        As[buf][r][k + 1] = f2bf(x4.y * s4.y + h4.y);
        As[buf][r][k + 2] = f2bf(x4.z * s4.z + h4.z);
        As[buf][r][k + 3] = f2bf(x4.w * s4.w + h4.w);
      } else {
        As[buf][r][k + 0] = f2bf(0.f);
        As[buf][r][k + 1] = f2bf(0.f);
        As[buf][r][k + 2] = f2bf(0.f);
        As[buf][r][k + 3] = f2bf(0.f);
      }
    }
  };

  v8f acc[NT] = {};

  stageA(0, 0);
  __syncthreads();

  for (int ks = 0; ks < KSTEPS; ++ks) {
    if (ks + 1 < KSTEPS) {
      __builtin_prefetch(&X[(size_t)(row0 + (t >> 1)) * K_TOT + ks * 32 + 32], 0, 1);
      stageA(ks + 1, (ks + 1) & 1);   // overlap next-tile staging with WMMAs
    }
    const int buf = ks & 1;
    const int k0 = ks * 32;

    // A fragment: lane<16 -> K {0..15}; lane>=16 -> K {16..31}
    Frag a;
    a.q[0] = *(const uint4*)&As[buf][mrow][khalf * 16];
    a.q[1] = *(const uint4*)&As[buf][mrow][khalf * 16 + 8];

    // B fragments: register double-buffer so DS loads overlap WMMA
    Frag bcur, bnxt;
    bcur.q[0] = *(const uint4*)&Bs[coln][k0 + khalf * 16];
    bcur.q[1] = *(const uint4*)&Bs[coln][k0 + khalf * 16 + 8];
#pragma unroll
    for (int nt = 0; nt < NT; ++nt) {
      if (nt + 1 < NT) {
        const int col = (nt + 1) * 16 + coln;
        bnxt.q[0] = *(const uint4*)&Bs[col][k0 + khalf * 16];
        bnxt.q[1] = *(const uint4*)&Bs[col][k0 + khalf * 16 + 8];
      }
      acc[nt] = __builtin_amdgcn_wmma_f32_16x16x32_bf16(
          false, a.v, false, bcur.v, (short)0, acc[nt], false, false);
      bcur = bnxt;
    }
    __syncthreads();
  }

  // ---- epilogue: C/D layout lane<16: M=r, N=lane; lane>=16: M=8+r, N=lane-16
  const int rsel = (lane >> 4) * 8;
#pragma unroll
  for (int nt = 0; nt < NT; ++nt) {
    const int col = nt * 16 + coln;
    float bv = HAS_BIAS ? bias[col] : 0.f;
#pragma unroll
    for (int r = 0; r < 8; ++r) {
      float v = acc[nt][r] + bv;
      if (LEAKY) v = leaky(v);
      Y[(size_t)(row0 + (wave << 4) + rsel + r) * NOUT + col] = v;
    }
  }
}

// ---------------- GCN scatter ----------------
template <int F>
__global__ __launch_bounds__(256)
void init_accum_kernel(float* __restrict__ out, const float* __restrict__ h,
                       const float* __restrict__ dinv, int n) {
  long long idx = (long long)blockIdx.x * 256 + threadIdx.x;
  if (idx >= (long long)n * F) return;
  int i = (int)(idx / F);
  float d = dinv[i];
  out[idx] = h[idx] * d * d;   // self-loop term
}

template <int F>
__global__ __launch_bounds__(256)
void edge_scatter_kernel(float* __restrict__ out, const float* __restrict__ h,
                         const float* __restrict__ dinv,
                         const int* __restrict__ src, const int* __restrict__ dst, int E) {
  constexpr int CH = F / 4;
  long long idx = (long long)blockIdx.x * 256 + threadIdx.x;
  if (idx >= (long long)E * CH) return;
  int e = (int)(idx / CH);
  int c = (int)(idx % CH);
  int s = src[e], d = dst[e];
  float wgt = dinv[s] * dinv[d];
  const float4 hv = *reinterpret_cast<const float4*>(&h[(size_t)s * F + c * 4]);
  float* o = &out[(size_t)d * F + c * 4];
  atomicAdd(o + 0, hv.x * wgt);
  atomicAdd(o + 1, hv.y * wgt);
  atomicAdd(o + 2, hv.z * wgt);
  atomicAdd(o + 3, hv.w * wgt);
}

template <int F>
__global__ __launch_bounds__(256)
void bias_leaky_kernel(float* __restrict__ x, const float* __restrict__ bias, int n) {
  long long idx = (long long)blockIdx.x * 256 + threadIdx.x;
  if (idx >= (long long)n * F) return;
  float v = x[idx] + bias[(int)(idx % F)];
  x[idx] = leaky(v);
}

// ---------------- depthwise 5x5 conv (SAME, zero pad), +bias, leaky ----------------
__global__ __launch_bounds__(256)
void dwconv_kernel(const float* __restrict__ in, const float* __restrict__ wdw,
                   const float* __restrict__ db, float* __restrict__ out) {
  __shared__ float sw[COUT * 25];
  int t = threadIdx.x;
  for (int i = t; i < COUT * 25; i += 256) sw[i] = wdw[i];
  __syncthreads();
  int idx = blockIdx.x * 256 + t;
  if (idx >= NN * COUT) return;
  int ch = idx & (COUT - 1);
  int pix = idx >> 6;
  int py = pix / WW, px = pix % WW;
  float acc = db[ch];
#pragma unroll
  for (int ky = 0; ky < 5; ++ky) {
    int yy = py + ky - 2;
    if (yy < 0 || yy >= HH) continue;
#pragma unroll
    for (int kx = 0; kx < 5; ++kx) {
      int xx = px + kx - 2;
      if (xx < 0 || xx >= WW) continue;
      acc += in[((size_t)(yy * WW + xx) << 6) + ch] * sw[ch * 25 + ky * 5 + kx];
    }
  }
  out[idx] = leaky(acc);
}

// ---------------- head: concat GEMM (128x16) + softmax ----------------
__global__ __launch_bounds__(256)
void head_kernel(const float* __restrict__ g, const float* __restrict__ cnn,
                 const float* __restrict__ lw, const float* __restrict__ lb,
                 float* __restrict__ out) {
  __shared__ float slw[HIDE * NCLS];
  __shared__ float slb[NCLS];
  int t = threadIdx.x;
  for (int i = t; i < HIDE * NCLS; i += 256) slw[i] = lw[i];
  if (t < NCLS) slb[t] = lb[t];
  __syncthreads();
  int p = blockIdx.x * 256 + t;
  if (p >= NN) return;
  float acc[NCLS];
#pragma unroll
  for (int j = 0; j < NCLS; ++j) acc[j] = slb[j];
  for (int f = 0; f < COUT; ++f) {
    float gv = g[(size_t)p * COUT + f];
#pragma unroll
    for (int j = 0; j < NCLS; ++j) acc[j] += gv * slw[f * NCLS + j];
  }
  for (int f = 0; f < COUT; ++f) {
    float cv = cnn[(size_t)p * COUT + f];
#pragma unroll
    for (int j = 0; j < NCLS; ++j) acc[j] += cv * slw[(COUT + f) * NCLS + j];
  }
  float m = acc[0];
#pragma unroll
  for (int j = 1; j < NCLS; ++j) m = fmaxf(m, acc[j]);
  float s = 0.f;
#pragma unroll
  for (int j = 0; j < NCLS; ++j) { acc[j] = __expf(acc[j] - m); s += acc[j]; }
  float inv = 1.0f / s;
#pragma unroll
  for (int j = 0; j < NCLS; ++j) out[(size_t)p * NCLS + j] = acc[j] * inv;
}

// ---------------- host-side orchestration ----------------
static inline unsigned cdiv(long long a, long long b) { return (unsigned)((a + b - 1) / b); }

static void bn_stats(const float* in, int Cn, float* ssum, float* ssq,
                     const float* g, const float* b, float* scale, float* shift,
                     hipStream_t st) {
  zero_kernel<<<2, 256, 0, st>>>(ssum, 512);  // sum(256) + sq(256) contiguous
  colstats_kernel<<<1024, 256, 0, st>>>(in, ssum, ssq, Cn, NN);
  finalize_affine_kernel<<<1, 256, 0, st>>>(ssum, ssq, g, b, scale, shift, Cn, 1.0f / NN);
}

extern "C" void kernel_launch(void* const* d_in, const int* in_sizes, int n_in,
                              void* d_out, int out_size, void* d_ws, size_t ws_size,
                              hipStream_t stream) {
  const float* x      = (const float*)d_in[0];
  const int*   edge   = (const int*)d_in[1];
  const int    E      = in_sizes[1] / 2;
  const int*   esrc   = edge;
  const int*   edst   = edge + E;
  const float* dn_bn1_g = (const float*)d_in[2];
  const float* dn_bn1_b = (const float*)d_in[3];
  const float* dn_w1    = (const float*)d_in[4];
  const float* dn_b1    = (const float*)d_in[5];
  const float* dn_bn2_g = (const float*)d_in[6];
  const float* dn_bn2_b = (const float*)d_in[7];
  const float* dn_w2    = (const float*)d_in[8];
  const float* dn_b2    = (const float*)d_in[9];
  const float* cnn_bn_g = (const float*)d_in[10];
  const float* cnn_bn_b = (const float*)d_in[11];
  const float* cnn_pw   = (const float*)d_in[12];
  const float* cnn_dw   = (const float*)d_in[13];
  const float* cnn_db   = (const float*)d_in[14];
  const float* g_bn_g[3] = {(const float*)d_in[15], (const float*)d_in[19], (const float*)d_in[23]};
  const float* g_bn_b[3] = {(const float*)d_in[16], (const float*)d_in[20], (const float*)d_in[24]};
  const float* g_w[3]    = {(const float*)d_in[17], (const float*)d_in[21], (const float*)d_in[25]};
  const float* g_b[3]    = {(const float*)d_in[18], (const float*)d_in[22], (const float*)d_in[26]};
  const float* lin_w    = (const float*)d_in[27];
  const float* lin_b    = (const float*)d_in[28];
  float* out = (float*)d_out;

  // workspace layout (floats)
  float* wsf      = (float*)d_ws;
  float* ssum     = wsf;                  // 256
  float* ssq      = wsf + 256;            // 256
  float* bscale   = wsf + 512;            // 256
  float* bshift   = wsf + 768;            // 256
  float* dinv     = wsf + 1024;           // N
  float* bufA     = dinv + NN;            // [N,128]
  float* bufB     = bufA + (size_t)NN * HIDE;
  float* bufC     = bufB + (size_t)NN * HIDE;
  float* bufD     = bufC + (size_t)NN * HIDE;
  float* c64a     = bufD + (size_t)NN * HIDE;  // [N,64]
  float* c64b     = c64a + (size_t)NN * COUT;  // [N,64]

  const dim3 blk(256);
  const dim3 gemm_grid(NN / 128);

  // ---- denoise layer 1: BN(x,200) -> GEMM 200x128 + bias + leaky -> bufA
  bn_stats(x, CC, ssum, ssq, dn_bn1_g, dn_bn1_b, bscale, bshift, stream);
  gemm_bn_wmma<CC, HIDE, true, true><<<gemm_grid, blk, 0, stream>>>(
      x, bscale, bshift, dn_w1, dn_b1, bufA);

  // ---- denoise layer 2: BN(128) -> GEMM 128x128 + bias + leaky -> bufB (clean)
  bn_stats(bufA, HIDE, ssum, ssq, dn_bn2_g, dn_bn2_b, bscale, bshift, stream);
  gemm_bn_wmma<HIDE, HIDE, true, true><<<gemm_grid, blk, 0, stream>>>(
      bufA, bscale, bshift, dn_w2, dn_b2, bufB);

  // ---- CNN branch: BN(128) -> GEMM 128x64 (no bias) + leaky -> c64a -> dwconv -> c64b
  bn_stats(bufB, HIDE, ssum, ssq, cnn_bn_g, cnn_bn_b, bscale, bshift, stream);
  gemm_bn_wmma<HIDE, COUT, true, false><<<gemm_grid, blk, 0, stream>>>(
      bufB, bscale, bshift, cnn_pw, nullptr, c64a);
  dwconv_kernel<<<cdiv((long long)NN * COUT, 256), blk, 0, stream>>>(c64a, cnn_dw, cnn_db, c64b);

  // ---- degrees: deg = 1 (self loop) + in-edges; dinv = rsqrt(deg)
  fill_kernel<<<cdiv(NN, 256), blk, 0, stream>>>(dinv, 1.0f, NN);
  edge_deg_kernel<<<cdiv(E, 256), blk, 0, stream>>>(dinv, edst, E);
  rsqrt_inplace_kernel<<<cdiv(NN, 256), blk, 0, stream>>>(dinv, NN);

  // ---- GCN layer 1 (128->128): in bufB, out bufD
  zero_kernel<<<2, 256, 0, stream>>>(ssum, 512);
  rownorm_stats_kernel<<<4096, 128, 0, stream>>>(bufB, bufA, ssum, ssq, NN);
  finalize_affine_kernel<<<1, 256, 0, stream>>>(ssum, ssq, g_bn_g[0], g_bn_b[0], bscale, bshift, HIDE, 1.0f / NN);
  gemm_bn_wmma<HIDE, HIDE, false, false><<<gemm_grid, blk, 0, stream>>>(
      bufA, bscale, bshift, g_w[0], nullptr, bufC);
  init_accum_kernel<HIDE><<<cdiv((long long)NN * HIDE, 256), blk, 0, stream>>>(bufD, bufC, dinv, NN);
  edge_scatter_kernel<HIDE><<<cdiv((long long)E * (HIDE / 4), 256), blk, 0, stream>>>(
      bufD, bufC, dinv, esrc, edst, E);
  bias_leaky_kernel<HIDE><<<cdiv((long long)NN * HIDE, 256), blk, 0, stream>>>(bufD, g_b[0], NN);

  // ---- GCN layer 2 (128->128): in bufD, out bufB
  zero_kernel<<<2, 256, 0, stream>>>(ssum, 512);
  rownorm_stats_kernel<<<4096, 128, 0, stream>>>(bufD, bufA, ssum, ssq, NN);
  finalize_affine_kernel<<<1, 256, 0, stream>>>(ssum, ssq, g_bn_g[1], g_bn_b[1], bscale, bshift, HIDE, 1.0f / NN);
  gemm_bn_wmma<HIDE, HIDE, false, false><<<gemm_grid, blk, 0, stream>>>(
      bufA, bscale, bshift, g_w[1], nullptr, bufC);
  init_accum_kernel<HIDE><<<cdiv((long long)NN * HIDE, 256), blk, 0, stream>>>(bufB, bufC, dinv, NN);
  edge_scatter_kernel<HIDE><<<cdiv((long long)E * (HIDE / 4), 256), blk, 0, stream>>>(
      bufB, bufC, dinv, esrc, edst, E);
  bias_leaky_kernel<HIDE><<<cdiv((long long)NN * HIDE, 256), blk, 0, stream>>>(bufB, g_b[1], NN);

  // ---- GCN layer 3 (128->64): in bufB, out bufD (as [N,64])
  zero_kernel<<<2, 256, 0, stream>>>(ssum, 512);
  rownorm_stats_kernel<<<4096, 128, 0, stream>>>(bufB, bufA, ssum, ssq, NN);
  finalize_affine_kernel<<<1, 256, 0, stream>>>(ssum, ssq, g_bn_g[2], g_bn_b[2], bscale, bshift, HIDE, 1.0f / NN);
  gemm_bn_wmma<HIDE, COUT, false, false><<<gemm_grid, blk, 0, stream>>>(
      bufA, bscale, bshift, g_w[2], nullptr, bufC);
  init_accum_kernel<COUT><<<cdiv((long long)NN * COUT, 256), blk, 0, stream>>>(bufD, bufC, dinv, NN);
  edge_scatter_kernel<COUT><<<cdiv((long long)E * (COUT / 4), 256), blk, 0, stream>>>(
      bufD, bufC, dinv, esrc, edst, E);
  bias_leaky_kernel<COUT><<<cdiv((long long)NN * COUT, 256), blk, 0, stream>>>(bufD, g_b[2], NN);

  // ---- head: [g | cnn_res] @ lin_w + lin_b -> softmax
  head_kernel<<<cdiv(NN, 256), blk, 0, stream>>>(bufD, c64b, lin_w, lin_b, out);
}
